// DepthAwareCrossAttention_48215302865352
// MI455X (gfx1250) — matrix operand order repaired and checked by reading, compiled, not run on hardware
//
#include <hip/hip_runtime.h>

typedef __attribute__((ext_vector_type(16))) _Float16 v16h;
typedef __attribute__((ext_vector_type(8)))  _Float16 v8h;
typedef __attribute__((ext_vector_type(8)))  float    v8f;

union AFrag { v16h v; v8h h[2]; };

#define WMMA_F16(a, b, c) \
  __builtin_amdgcn_wmma_f32_16x16x32_f16(false, (a), false, (b), (short)0, (c), false, false)

static constexpr int kC   = 256;   // C1 == C2
static constexpr int kH1  = 128;
static constexpr int kW1  = 128;
static constexpr int kH2  = 64;
static constexpr int kW2  = 200;
static constexpr int kPix = kH1 * kW1;   // 16384
static constexpr int kMQ  = kW2 * kH1;   // 25600 query rows
static constexpr int kMK  = kW2 * kH2;   // 12800 key/value rows
static constexpr int kNH  = 8;

__device__ __forceinline__ int iclamp(int v, int lo, int hi) {
  return v < lo ? lo : (v > hi ? hi : v);
}

// polar_coords from the reference, for H=W=128 (cx=cy=64), h_out=128, w_out=200
__device__ __forceinline__ void polar(int t, int bw, const float* fov, const float* rots,
                                      int cam, float& y, float& x) {
  float f = fov[cam];
  float c =  rots[cam * 9 + 3];   // rot[0,0] = R[1][0]
  float s = -rots[cam * 9 + 0];   // rot[1,0] = -R[0][0]
  float tt  = (float)bw * (1.0f / 199.0f);
  float ang = f * (tt - 0.5f);
  float cs = __cosf(ang), sn = __sinf(ang);
  float ca =  c * cs + s * sn;
  float sa = -s * cs + c * sn;
  float rad = ((float)t * (1.0f / 127.0f)) * 90.50966799187809f;  // sqrt(64^2+64^2)
  x = fminf(fmaxf(64.0f + rad * ca, 0.0f), 127.0f);
  y = fminf(fmaxf(64.0f - rad * sa, 0.0f), 127.0f);
}

// ---------------------------------------------------------------------------
// Weight prep: Wc[g] = in_w[g]·{q,k,v}_w  (f16),  bc[g] = in_w[g]·{q,k,v}_b + in_b[g]
// ---------------------------------------------------------------------------
__global__ void prep_weights(const float* __restrict__ in_w, const float* __restrict__ in_b,
                             const float* __restrict__ qw, const float* __restrict__ qb,
                             const float* __restrict__ kw, const float* __restrict__ kb,
                             const float* __restrict__ vw, const float* __restrict__ vb,
                             _Float16* __restrict__ Wc, float* __restrict__ bc) {
  int g = blockIdx.y;       // 0:q 1:k 2:v
  int o = blockIdx.x;       // output channel
  int k = threadIdx.x;      // input channel
  const float* sw = (g == 0) ? qw : (g == 1) ? kw : vw;
  const float* sb = (g == 0) ? qb : (g == 1) ? kb : vb;
  const float* row = in_w + (size_t)(g * kC + o) * kC;
  float acc = 0.f;
  for (int j = 0; j < kC; ++j) acc += row[j] * sw[j * kC + k];
  Wc[(size_t)g * kC * kC + (size_t)o * kC + k] = (_Float16)acc;
  if (k == 0) {
    float ba = in_b[g * kC + o];
    for (int j = 0; j < kC; ++j) ba += row[j] * sb[j];
    bc[g * kC + o] = ba;
  }
}

__global__ void f32_to_f16(const float* __restrict__ src, _Float16* __restrict__ dst, int n) {
  int i = blockIdx.x * blockDim.x + threadIdx.x;
  if (i < n) dst[i] = (_Float16)src[i];
}

// ---------------------------------------------------------------------------
// extract_canvas + pos_a fold + f16 convert.  Row layout: (bw*128 + t, c)
// ---------------------------------------------------------------------------
__global__ void extract_q(const float* __restrict__ aw, const float* __restrict__ fov,
                          const float* __restrict__ rots, const float* __restrict__ pos_a,
                          _Float16* __restrict__ Xq, int cam) {
  int pix = blockIdx.x;           // bw*128 + t
  int bw = pix >> 7, t = pix & 127;
  float y, x;
  polar(t, bw, fov, rots, cam, y, x);
  float x0 = floorf(x), y0 = floorf(y);
  float wx = x - x0, wy = y - y0;
  int x0i = iclamp((int)x0, 0, 127), x1i = iclamp((int)x0 + 1, 0, 127);
  int y0i = iclamp((int)y0, 0, 127), y1i = iclamp((int)y0 + 1, 0, 127);
  int c = threadIdx.x;
  const float* ch = aw + (size_t)c * kPix;
  float v00 = ch[y0i * 128 + x0i], v01 = ch[y0i * 128 + x1i];
  float v10 = ch[y1i * 128 + x0i], v11 = ch[y1i * 128 + x1i];
  float v = (1.f - wy) * ((1.f - wx) * v00 + wx * v01) +
            wy * ((1.f - wx) * v10 + wx * v11);
  Xq[(size_t)pix * kC + c] = (_Float16)(v + pos_a[t * kC + c]);
}

// b_j transpose + pos_b fold + f16 convert.  Row layout: (bw*64 + s, c)
__global__ void stage_kv(const float* __restrict__ bsrc, const float* __restrict__ pos_b,
                         _Float16* __restrict__ Xk) {
  int pix = blockIdx.x;           // bw*64 + s
  int bw = pix >> 6, s = pix & 63;
  int c = threadIdx.x;
  float v = bsrc[(size_t)c * (kH2 * kW2) + s * kW2 + bw] + pos_b[s * kC + c];
  Xk[(size_t)pix * kC + c] = (_Float16)v;
}

// ---------------------------------------------------------------------------
// WMMA GEMM: out[M,256] = (A[M,256] · W[256,256]^T + bias) * scale
// block = 8 waves; wave tile 16(M)x64(N); grid = (M/128, 4)
// ---------------------------------------------------------------------------
template <bool OUT_F16>
__global__ void gemm_wmma(const _Float16* __restrict__ A, const _Float16* __restrict__ W,
                          const float* __restrict__ bias, void* __restrict__ outp,
                          float scale) {
  int wave = threadIdx.x >> 5;
  int lane = threadIdx.x & 31;
  int half = lane >> 4, lm = lane & 15;
  int m0 = blockIdx.x * 128 + wave * 16;
  int n0 = blockIdx.y * 64;

  v8f c0 = {}, c1 = {}, c2 = {}, c3 = {};
  const _Float16* arow = A + (size_t)(m0 + lm) * kC + half * 16;
  const _Float16* wrow = W + (size_t)(n0 + lm) * kC + half * 16;
#pragma unroll
  for (int k0 = 0; k0 < kC; k0 += 32) {
    AFrag a;
    a.h[0] = *(const v8h*)(arow + k0);
    a.h[1] = *(const v8h*)(arow + k0 + 8);
    v16h b0 = *(const v16h*)(wrow + k0);
    v16h b1 = *(const v16h*)(wrow + k0 + 16 * kC);
    v16h b2 = *(const v16h*)(wrow + k0 + 32 * kC);
    v16h b3 = *(const v16h*)(wrow + k0 + 48 * kC);
    c0 = WMMA_F16(a.v, b0, c0);
    c1 = WMMA_F16(a.v, b1, c1);
    c2 = WMMA_F16(a.v, b2, c2);
    c3 = WMMA_F16(a.v, b3, c3);
  }
  v8f cc[4] = {c0, c1, c2, c3};
#pragma unroll
  for (int nt = 0; nt < 4; ++nt) {
    int ncol = n0 + nt * 16 + lm;
    float bv = bias[ncol];
#pragma unroll
    for (int r = 0; r < 8; ++r) {
      int m = m0 + r + half * 8;
      float val = (cc[nt][r] + bv) * scale;
      if (OUT_F16)
        ((_Float16*)outp)[(size_t)m * kC + ncol] = (_Float16)val;
      else
        ((float*)outp)[(size_t)m * kC + ncol] = val;
    }
  }
}

// ---------------------------------------------------------------------------
// Attention: per (b,h): logits = Qh·Khᵀ (K-contraction = hd = 32, one WMMA/tile),
// row softmax (S=64), out = P·Vh (two 32-step WMMAs).  Scale pre-folded into Q.
// ---------------------------------------------------------------------------
__global__ void attn_wmma(const _Float16* __restrict__ Q, const _Float16* __restrict__ K,
                          const _Float16* __restrict__ V, _Float16* __restrict__ O) {
  __shared__ _Float16 Pl[128][80];   // stride 80 halves = 160B -> 16B-aligned v8h rows
  __shared__ _Float16 Vl[64][32];
  int b = blockIdx.x, h = blockIdx.y;
  int wave = threadIdx.x >> 5, lane = threadIdx.x & 31;
  int half = lane >> 4, lm = lane & 15;
  int m0 = wave * 16;
  int hc = h * 32;
  size_t q0 = (size_t)b * 128;
  size_t k0 = (size_t)b * 64;

  // stage V head tile coalesced (2048 halves / 256 threads)
#pragma unroll
  for (int e = 0; e < 8; ++e) {
    int lin = threadIdx.x + 256 * e;
    int s = lin >> 5, d = lin & 31;
    Vl[s][d] = V[(k0 + s) * kC + hc + d];
  }

  // logits for this wave's 16-row T strip, all 64 S columns (4 tiles)
  AFrag a;
  const _Float16* ap = Q + (q0 + m0 + lm) * kC + hc + half * 16;
  a.h[0] = *(const v8h*)ap;
  a.h[1] = *(const v8h*)(ap + 8);
  v8f l0 = {}, l1 = {}, l2 = {}, l3 = {};
  {
    const _Float16* kp = K + (k0 + lm) * kC + hc + half * 16;
    v16h b0 = *(const v16h*)(kp);
    v16h b1 = *(const v16h*)(kp + 16 * kC);
    v16h b2 = *(const v16h*)(kp + 32 * kC);
    v16h b3 = *(const v16h*)(kp + 48 * kC);
    l0 = WMMA_F16(a.v, b0, l0);
    l1 = WMMA_F16(a.v, b1, l1);
    l2 = WMMA_F16(a.v, b2, l2);
    l3 = WMMA_F16(a.v, b3, l3);
  }
  v8f L[4] = {l0, l1, l2, l3};
  // per-row softmax: 4 tile regs + 16-lane xor-shuffle tree (wave32)
#pragma unroll
  for (int r = 0; r < 8; ++r) {
    float mx = fmaxf(fmaxf(L[0][r], L[1][r]), fmaxf(L[2][r], L[3][r]));
#pragma unroll
    for (int off = 1; off < 16; off <<= 1) mx = fmaxf(mx, __shfl_xor(mx, off, 32));
    float e0 = __expf(L[0][r] - mx);
    float e1 = __expf(L[1][r] - mx);
    float e2 = __expf(L[2][r] - mx);
    float e3 = __expf(L[3][r] - mx);
    float sm = e0 + e1 + e2 + e3;
#pragma unroll
    for (int off = 1; off < 16; off <<= 1) sm += __shfl_xor(sm, off, 32);
    float inv = 1.0f / sm;
    int m = m0 + r + half * 8;
    Pl[m][ 0 + lm] = (_Float16)(e0 * inv);
    Pl[m][16 + lm] = (_Float16)(e1 * inv);
    Pl[m][32 + lm] = (_Float16)(e2 * inv);
    Pl[m][48 + lm] = (_Float16)(e3 * inv);
  }
  __syncthreads();

  // out = P(16x64) @ V(64x32): 2 k-steps x 2 n-tiles
  v8f o0 = {}, o1 = {};
#pragma unroll
  for (int kt = 0; kt < 2; ++kt) {
    AFrag p;
    const _Float16* pp = &Pl[m0 + lm][kt * 32 + half * 16];
    p.h[0] = *(const v8h*)pp;
    p.h[1] = *(const v8h*)(pp + 8);
    v16h bv0, bv1;
#pragma unroll
    for (int j = 0; j < 16; ++j) {
      int s = kt * 32 + half * 16 + j;
      bv0[j] = Vl[s][ 0 + lm];
      bv1[j] = Vl[s][16 + lm];
    }
    o0 = WMMA_F16(p.v, bv0, o0);
    o1 = WMMA_F16(p.v, bv1, o1);
  }
#pragma unroll
  for (int r = 0; r < 8; ++r) {
    int m = m0 + r + half * 8;
    O[(q0 + m) * kC + hc +  0 + lm] = (_Float16)o0[r];
    O[(q0 + m) * kC + hc + 16 + lm] = (_Float16)o1[r];
  }
}

// ---------------------------------------------------------------------------
// restore_canvas: rounded scatter-add (atomics) + count
// ---------------------------------------------------------------------------
__global__ void scatter_restore(const float* __restrict__ Y, const float* __restrict__ fov,
                                const float* __restrict__ rots, float* __restrict__ acc,
                                float* __restrict__ cnt, int cam) {
  int pix = blockIdx.x;           // bw*128 + t
  int bw = pix >> 7, t = pix & 127;
  float y, x;
  polar(t, bw, fov, rots, cam, y, x);
  int xi = iclamp((int)rintf(x), 0, 127);
  int yi = iclamp((int)rintf(y), 0, 127);
  int p = yi * 128 + xi;
  int c = threadIdx.x;
  atomicAdd(&acc[(size_t)c * kPix + p], Y[(size_t)pix * kC + c]);
  if (c == 0) atomicAdd(&cnt[p], 1.0f);
}

// a_i += enh ; if last camera: out = a_i + enh
__global__ void finalize_restore(float* __restrict__ aw, const float* __restrict__ acc,
                                 const float* __restrict__ cnt, float* __restrict__ outp) {
  int idx = blockIdx.x * 256 + threadIdx.x;   // 0 .. 256*16384-1
  float ct = cnt[idx & (kPix - 1)];
  ct = (ct == 0.f) ? 1.f : ct;
  float v = acc[idx] / ct;
  float nv = aw[idx] + v;
  aw[idx] = nv;
  if (outp) outp[idx] = nv + v;
}

// ---------------------------------------------------------------------------
extern "C" void kernel_launch(void* const* d_in, const int* in_sizes, int n_in,
                              void* d_out, int out_size, void* d_ws, size_t ws_size,
                              hipStream_t stream) {
  (void)in_sizes; (void)n_in; (void)out_size; (void)ws_size;
  const float* a     = (const float*)d_in[0];
  const float* bb    = (const float*)d_in[1];
  const float* fov   = (const float*)d_in[2];
  const float* rots  = (const float*)d_in[3];
  const float* qw    = (const float*)d_in[7];
  const float* qb    = (const float*)d_in[8];
  const float* kw    = (const float*)d_in[9];
  const float* kb    = (const float*)d_in[10];
  const float* vw    = (const float*)d_in[11];
  const float* vb    = (const float*)d_in[12];
  const float* pos_a = (const float*)d_in[13];
  const float* pos_b = (const float*)d_in[14];
  const float* in_w  = (const float*)d_in[15];
  const float* in_b  = (const float*)d_in[16];
  const float* ow    = (const float*)d_in[17];
  const float* ob    = (const float*)d_in[18];
  float* out = (float*)d_out;

  size_t off = 0;
  auto carve = [&](size_t bytes) -> void* {
    void* p = (char*)d_ws + off;
    off += (bytes + 255) & ~(size_t)255;
    return p;
  };
  float*    awork = (float*)carve((size_t)2 * kC * kPix * 4);
  _Float16* Wc    = (_Float16*)carve((size_t)3 * kC * kC * 2);
  _Float16* Wo    = (_Float16*)carve((size_t)kC * kC * 2);
  float*    bc    = (float*)carve((size_t)3 * kC * 4);
  _Float16* Xq    = (_Float16*)carve((size_t)kMQ * kC * 2);
  _Float16* Xk    = (_Float16*)carve((size_t)kMK * kC * 2);
  _Float16* Qb    = (_Float16*)carve((size_t)kMQ * kC * 2);
  _Float16* Kb    = (_Float16*)carve((size_t)kMK * kC * 2);
  _Float16* Vb    = (_Float16*)carve((size_t)kMK * kC * 2);
  _Float16* At    = (_Float16*)carve((size_t)kMQ * kC * 2);
  float*    Yb    = (float*)carve((size_t)kMQ * kC * 4);
  float*    acc   = (float*)carve((size_t)kC * kPix * 4);
  float*    cntb  = (float*)carve((size_t)kPix * 4);

  hipMemcpyAsync(awork, a, (size_t)2 * kC * kPix * 4, hipMemcpyDeviceToDevice, stream);
  prep_weights<<<dim3(kC, 3), kC, 0, stream>>>(in_w, in_b, qw, qb, kw, kb, vw, vb, Wc, bc);
  f32_to_f16<<<kC * kC / 256, 256, 0, stream>>>(ow, Wo, kC * kC);

  const float qscale = 0.17677669529663687f;   // 1/sqrt(hd=32), folded into Q
  for (int i = 0; i < 2; ++i) {
    float* aw = awork + (size_t)i * kC * kPix;
    const int cams[2] = {0, 3};
    for (int cj = 0; cj < 2; ++cj) {
      int cam = i * 4 + cams[cj];
      bool last = (cj == 1);
      extract_q<<<kMQ, kC, 0, stream>>>(aw, fov, rots, pos_a, Xq, cam);
      stage_kv<<<kMK, kC, 0, stream>>>(bb + (size_t)cam * kC * kH2 * kW2, pos_b, Xk);
      gemm_wmma<true><<<dim3(kMQ / 128, 4), 256, 0, stream>>>(Xq, Wc,               bc,          Qb, qscale);
      gemm_wmma<true><<<dim3(kMK / 128, 4), 256, 0, stream>>>(Xk, Wc + kC * kC,     bc + kC,     Kb, 1.0f);
      gemm_wmma<true><<<dim3(kMK / 128, 4), 256, 0, stream>>>(Xk, Wc + 2 * kC * kC, bc + 2 * kC, Vb, 1.0f);
      attn_wmma<<<dim3(kW2, kNH), 256, 0, stream>>>(Qb, Kb, Vb, At);
      gemm_wmma<false><<<dim3(kMQ / 128, 4), 256, 0, stream>>>(At, Wo, ob, Yb, 1.0f);
      hipMemsetAsync(acc, 0, (size_t)kC * kPix * 4, stream);
      hipMemsetAsync(cntb, 0, (size_t)kPix * 4, stream);
      scatter_restore<<<kMQ, kC, 0, stream>>>(Yb, fov, rots, acc, cntb, cam);
      finalize_restore<<<kC * kPix / 256, 256, 0, stream>>>(
          aw, acc, cntb, last ? (out + (size_t)i * kC * kPix) : (float*)nullptr);
    }
  }
}
